// UpSample_68719477130
// MI455X (gfx1250) — matrix operand order repaired and checked by reading, not compiled, not run on hardware
//
#include <hip/hip_runtime.h>

// ---------------------------------------------------------------------------
// MI455X (gfx1250) fused cross-attention:
//   out = Wskip@up + softmax((Wq@up)^T (Wk@down) / sqrt(C)) applied to (Wv@down)
// All GEMMs via v_wmma_f32_16x16x32_bf16 (wave32). B=8, C=512, N=4096, M=1024.
// ---------------------------------------------------------------------------

typedef __attribute__((ext_vector_type(8)))  float   v8f;
typedef __attribute__((ext_vector_type(16))) __bf16  v16bf;
typedef __attribute__((ext_vector_type(8)))  __bf16  v8bf;

#define WMMA_BF16(a, b, c) \
  __builtin_amdgcn_wmma_f32_16x16x32_bf16(false, (a), false, (b), (short)0, (c), false, false)

constexpr int Bb = 8;
constexpr int Cc = 512;
constexpr int Nn = 4096;
constexpr int Mm = 1024;

union AFrag { v16bf vv; v8bf h[2]; };

static __device__ inline v8f zero8() {
  v8f z;
#pragma unroll
  for (int i = 0; i < 8; ++i) z[i] = 0.0f;
  return z;
}

// -------------------------------- K0a: f32 -> bf16 copy ---------------------
__global__ void cvt_bf16_kernel(const float* __restrict__ in, __bf16* __restrict__ out, int n) {
  int i = blockIdx.x * blockDim.x + threadIdx.x;
  if (i < n) out[i] = (__bf16)in[i];
}

// ------------------------- K0b: [Cr, Nc] f32 -> [Nc, Cr] bf16 ---------------
__global__ void transpose_cvt_kernel(const float* __restrict__ in, __bf16* __restrict__ out,
                                     int Cr, int Nc) {
  __shared__ float tile[32][33];
  int b  = blockIdx.z;
  const float* ip = in  + (size_t)b * Cr * Nc;
  __bf16*      op = out + (size_t)b * Cr * Nc;
  int x = threadIdx.x & 31;
  int y = threadIdx.x >> 5;                 // 0..7
  int n0 = blockIdx.x * 32;
  int c0 = blockIdx.y * 32;
#pragma unroll
  for (int k = 0; k < 4; ++k)
    tile[y + k * 8][x] = ip[(size_t)(c0 + y + k * 8) * Nc + n0 + x];
  __syncthreads();
#pragma unroll
  for (int k = 0; k < 4; ++k)
    op[(size_t)(n0 + y + k * 8) * Cr + c0 + x] = (__bf16)tile[x][y + k * 8];
}

// ---------------- K1: NT GEMM  Out[i,j] = sum_k A[i,k]*B[j,k]  (bf16 out) ----
// A: [rows, K] bf16 row-major, B: [cols, K] bf16 row-major.
// Block = 8 waves; wave computes 16x64; block computes 128x64.
__global__ __launch_bounds__(256) void gemm_nt_bf16_kernel(
    const __bf16* __restrict__ A, const __bf16* __restrict__ B, __bf16* __restrict__ Out,
    int K, int ldo, long long strideA, long long strideB, long long strideO) {
  int b    = blockIdx.z;
  const __bf16* Ap = A + strideA * b;
  const __bf16* Bp = B + strideB * b;
  __bf16*       Op = Out + strideO * b;
  int lane = threadIdx.x & 31, wv = threadIdx.x >> 5;
  int l16 = lane & 15, hi = lane >> 4;
  int i0 = blockIdx.x * 128 + wv * 16;
  int j0 = blockIdx.y * 64;

  v8f acc[4];
#pragma unroll
  for (int t = 0; t < 4; ++t) acc[t] = zero8();

  for (int k = 0; k < K; k += 32) {
    AFrag af;
    const __bf16* arow = Ap + (size_t)(i0 + l16) * K + k;
    af.h[0] = *(const v8bf*)(arow + hi * 16);
    af.h[1] = *(const v8bf*)(arow + hi * 16 + 8);
#pragma unroll
    for (int t = 0; t < 4; ++t) {
      const __bf16* brow = Bp + (size_t)(j0 + t * 16 + l16) * K + k + hi * 16;
      acc[t] = WMMA_BF16(af.vv, *(const v16bf*)brow, acc[t]);
    }
  }
#pragma unroll
  for (int t = 0; t < 4; ++t)
#pragma unroll
    for (int r = 0; r < 8; ++r)
      Op[(size_t)(i0 + r + 8 * hi) * ldo + j0 + t * 16 + l16] = (__bf16)acc[t][r];
}

// ---------------- K2: fused q-proj + energy + softmax + (skip + attn@v^T) ----
// One block = 16 query rows (n0..n0+15) of one batch; 8 waves.
constexpr int QP = Cc + 8;   // qbuf pitch (bf16)
constexpr int EP = Mm + 1;   // ebuf pitch (f32)
constexpr int AP = Mm + 8;   // attn pitch (bf16)
constexpr int XP = Cc + 1;   // xbuf pitch (f32)
constexpr size_t SMEM_BYTES =
    (size_t)16 * QP * 2 + (size_t)16 * EP * 4 + (size_t)16 * AP * 2 +
    (size_t)16 * XP * 4 + (256 + 16 + 16) * 4;

__global__ __launch_bounds__(256) void attn_fused_kernel(
    const __bf16* __restrict__ upT,   // [B, N, C]
    const __bf16* __restrict__ kT,    // [B, M, C]
    const __bf16* __restrict__ vv,    // [B, C, M]
    const __bf16* __restrict__ Wq,    // [C, C] bf16 (row o, contiguous c)
    const __bf16* __restrict__ Wsk,   // [C, C] bf16
    float* __restrict__ out) {        // [B, C, N]
  extern __shared__ char smem[];
  __bf16* qbuf = (__bf16*)smem;                                  // [16][QP]
  float*  ebuf = (float*)(smem + (size_t)16 * QP * 2);           // [16][EP]
  __bf16* abuf = (__bf16*)((char*)ebuf + (size_t)16 * EP * 4);   // [16][AP]
  float*  xbuf = (float*)((char*)abuf + (size_t)16 * AP * 2);    // [16][XP]
  float*  pred = xbuf + (size_t)16 * XP;                         // [16][16]
  float*  rmax = pred + 256;                                     // [16]
  float*  rinv = rmax + 16;                                      // [16]

  int b  = blockIdx.y;
  int n0 = blockIdx.x * 16;
  int tid  = threadIdx.x;
  int lane = tid & 31, wv = tid >> 5;
  int l16 = lane & 15, hi = lane >> 4;

  const __bf16* up_b = upT + (size_t)b * Nn * Cc + (size_t)n0 * Cc;
  const __bf16* kT_b = kT + (size_t)b * Mm * Cc;
  const __bf16* v_b  = vv + (size_t)b * Cc * Mm;

  // ---- Phase A: q tile [16 x C] = upT_tile x Wq  (wave w owns 64 c-cols) ----
  {
    int c0w = wv * 64;
    v8f acc[4];
#pragma unroll
    for (int t = 0; t < 4; ++t) acc[t] = zero8();
    for (int k = 0; k < Cc; k += 32) {
      AFrag af;
      const __bf16* arow = up_b + (size_t)l16 * Cc + k;
      af.h[0] = *(const v8bf*)(arow + hi * 16);
      af.h[1] = *(const v8bf*)(arow + hi * 16 + 8);
#pragma unroll
      for (int t = 0; t < 4; ++t) {
        const __bf16* brow = Wq + (size_t)(c0w + t * 16 + l16) * Cc + k + hi * 16;
        acc[t] = WMMA_BF16(af.vv, *(const v16bf*)brow, acc[t]);
      }
    }
#pragma unroll
    for (int t = 0; t < 4; ++t)
#pragma unroll
      for (int r = 0; r < 8; ++r)
        qbuf[(r + 8 * hi) * QP + c0w + t * 16 + l16] = (__bf16)acc[t][r];
  }
  __syncthreads();

  // ---- Phase B: energy strip [16 x 1024] (wave w owns m in [w*128, w*128+128)) ----
  {
    AFrag qf[16];
#pragma unroll
    for (int kk = 0; kk < 16; ++kk) {   // hoist q A-fragments: reused 8x
      const __bf16* qrow = qbuf + l16 * QP + kk * 32;
      qf[kk].h[0] = *(const v8bf*)(qrow + hi * 16);
      qf[kk].h[1] = *(const v8bf*)(qrow + hi * 16 + 8);
    }
    const float escale = 0.044194173824159216f;   // 1/sqrt(512)
    for (int mt = 0; mt < 8; ++mt) {
      int m0 = wv * 128 + mt * 16;
      v8f acc = zero8();
#pragma unroll
      for (int kk = 0; kk < 16; ++kk) {
        const __bf16* brow = kT_b + (size_t)(m0 + l16) * Cc + kk * 32 + hi * 16;
        acc = WMMA_BF16(qf[kk].vv, *(const v16bf*)brow, acc);
      }
#pragma unroll
      for (int r = 0; r < 8; ++r)
        ebuf[(r + 8 * hi) * EP + m0 + l16] = acc[r] * escale;
    }
  }
  __syncthreads();

  // ---- Phase C: row softmax over m (unnormalized p -> abuf bf16, 1/sum -> rinv) ----
  {
    int row  = tid >> 4;   // 0..15
    int slot = tid & 15;   // 0..15
    float mx = -3.0e38f;
    for (int k = 0; k < 64; ++k)
      mx = fmaxf(mx, ebuf[row * EP + slot + k * 16]);
    pred[row * 16 + slot] = mx;
    __syncthreads();
    if (slot == 0) {
      float m2 = pred[row * 16];
#pragma unroll
      for (int j = 1; j < 16; ++j) m2 = fmaxf(m2, pred[row * 16 + j]);
      rmax[row] = m2;
    }
    __syncthreads();
    float rm = rmax[row];
    float s = 0.0f;
    for (int k = 0; k < 64; ++k) {
      int m = slot + k * 16;
      float p = __expf(ebuf[row * EP + m] - rm);
      abuf[row * AP + m] = (__bf16)p;
      s += p;
    }
    pred[row * 16 + slot] = s;
    __syncthreads();
    if (slot == 0) {
      float t = 0.0f;
#pragma unroll
      for (int j = 0; j < 16; ++j) t += pred[row * 16 + j];
      rinv[row] = 1.0f / t;
    }
  }
  __syncthreads();

  // ---- Phase D: x = attn @ v^T and fused skip = upT_tile x Wskip ----
  {
    int c0w = wv * 64;
    v8f sacc[4], xacc[4];
#pragma unroll
    for (int t = 0; t < 4; ++t) { sacc[t] = zero8(); xacc[t] = zero8(); }

    for (int k = 0; k < Cc; k += 32) {      // skip projection
      AFrag af;
      const __bf16* arow = up_b + (size_t)l16 * Cc + k;
      af.h[0] = *(const v8bf*)(arow + hi * 16);
      af.h[1] = *(const v8bf*)(arow + hi * 16 + 8);
#pragma unroll
      for (int t = 0; t < 4; ++t) {
        const __bf16* brow = Wsk + (size_t)(c0w + t * 16 + l16) * Cc + k + hi * 16;
        sacc[t] = WMMA_BF16(af.vv, *(const v16bf*)brow, sacc[t]);
      }
    }
    for (int mm = 0; mm < Mm; mm += 32) {   // attn @ v^T
      AFrag af;
      const __bf16* arow = abuf + l16 * AP + mm;
      af.h[0] = *(const v8bf*)(arow + hi * 16);
      af.h[1] = *(const v8bf*)(arow + hi * 16 + 8);
#pragma unroll
      for (int t = 0; t < 4; ++t) {
        const __bf16* brow = v_b + (size_t)(c0w + t * 16 + l16) * Mm + mm + hi * 16;
        xacc[t] = WMMA_BF16(af.vv, *(const v16bf*)brow, xacc[t]);
      }
    }
#pragma unroll
    for (int r = 0; r < 8; ++r) {
      float ri = rinv[r + 8 * hi];
#pragma unroll
      for (int t = 0; t < 4; ++t)
        xbuf[(r + 8 * hi) * XP + c0w + t * 16 + l16] = sacc[t][r] + ri * xacc[t][r];
    }
  }
  __syncthreads();

  // ---- Phase E: coalesced transpose-store to out[b][c][n0+j] ----
  {
    int j  = tid & 15;    // n offset: lanes 0..15 -> consecutive addresses
    int cb = tid >> 4;    // 0..15
    float* ob = out + (size_t)b * Cc * Nn + n0;
    for (int cc = 0; cc < 32; ++cc) {
      int c = cc * 16 + cb;
      ob[(size_t)c * Nn + j] = xbuf[j * XP + c];
    }
  }
}

// ---------------------------------------------------------------------------
extern "C" void kernel_launch(void* const* d_in, const int* in_sizes, int n_in,
                              void* d_out, int out_size, void* d_ws, size_t ws_size,
                              hipStream_t stream) {
  const float* pcd_up   = (const float*)d_in[0];
  const float* pcd_down = (const float*)d_in[1];
  const float* Wq   = (const float*)d_in[2];
  const float* Wk   = (const float*)d_in[3];
  const float* Wv   = (const float*)d_in[4];
  const float* Wsk  = (const float*)d_in[5];
  float* out = (float*)d_out;

  char* ws = (char*)d_ws;
  auto carve = [&](size_t bytes) -> char* {
    char* p = ws;
    ws += (bytes + 255) & ~(size_t)255;
    return p;
  };
  __bf16* upT   = (__bf16*)carve((size_t)Bb * Nn * Cc * 2);
  __bf16* downT = (__bf16*)carve((size_t)Bb * Mm * Cc * 2);
  __bf16* wq_b  = (__bf16*)carve((size_t)Cc * Cc * 2);
  __bf16* wk_b  = (__bf16*)carve((size_t)Cc * Cc * 2);
  __bf16* wv_b  = (__bf16*)carve((size_t)Cc * Cc * 2);
  __bf16* ws_b  = (__bf16*)carve((size_t)Cc * Cc * 2);
  __bf16* kTb   = (__bf16*)carve((size_t)Bb * Mm * Cc * 2);
  __bf16* vb    = (__bf16*)carve((size_t)Bb * Cc * Mm * 2);

  // 1) weights -> bf16
  int wn = Cc * Cc;
  cvt_bf16_kernel<<<dim3(wn / 256), 256, 0, stream>>>(Wq,  wq_b, wn);
  cvt_bf16_kernel<<<dim3(wn / 256), 256, 0, stream>>>(Wk,  wk_b, wn);
  cvt_bf16_kernel<<<dim3(wn / 256), 256, 0, stream>>>(Wv,  wv_b, wn);
  cvt_bf16_kernel<<<dim3(wn / 256), 256, 0, stream>>>(Wsk, ws_b, wn);

  // 2) transpose-convert inputs: [B,C,N] f32 -> [B,N,C] bf16
  transpose_cvt_kernel<<<dim3(Nn / 32, Cc / 32, Bb), 256, 0, stream>>>(pcd_up, upT, Cc, Nn);
  transpose_cvt_kernel<<<dim3(Mm / 32, Cc / 32, Bb), 256, 0, stream>>>(pcd_down, downT, Cc, Mm);

  // 3) kT[b,m,o] = sum_c downT[b,m,c] * Wk[o,c]   -> [B, M, C] bf16
  gemm_nt_bf16_kernel<<<dim3(Mm / 128, Cc / 64, Bb), 256, 0, stream>>>(
      downT, wk_b, kTb, Cc, Cc, (long long)Mm * Cc, 0, (long long)Mm * Cc);
  //    v[b,o,m] = sum_c Wv[o,c] * downT[b,m,c]    -> [B, C, M] bf16
  gemm_nt_bf16_kernel<<<dim3(Cc / 128, Mm / 64, Bb), 256, 0, stream>>>(
      wv_b, downT, vb, Cc, Mm, 0, (long long)Mm * Cc, (long long)Cc * Mm);

  // 4) fused q-proj + energy + softmax + (skip + attn@v^T) + transpose-store
  attn_fused_kernel<<<dim3(Nn / 16, Bb), 256, (unsigned)SMEM_BYTES, stream>>>(
      upT, kTb, vb, wq_b, ws_b, out);
}
